// MultiInferModel_20933670601275
// MI455X (gfx1250) — compile-verified, matrix operand order checked by reading, not run on hardware
//
#include <hip/hip_runtime.h>
#include <hip/hip_bf16.h>

// Problem constants (match reference)
#define BB 2
#define LL 128
#define HH 768
#define CC 6
#define K1 1536                 // 2H (feature width)
#define KF 1554                 // 2H + 3C (concat width)
#define KP 1568                 // K padded to 49*32 for WMMA k-steps
#define MROWS (BB*LL*LL)        // 32768 pair-rows
#define NTILES (K1/16)          // 96 n-tiles of feat GEMM
#define KTILES (KP/32)          // 49 k-steps (feat GEMM)
#define KTILES_C (K1/32)        // 48 k-steps (logits GEMM)

typedef __attribute__((ext_vector_type(16))) __bf16         v16bf;
typedef __attribute__((ext_vector_type(8)))  float          v8f;
typedef __attribute__((ext_vector_type(8)))  unsigned short ushort8;

union Frag { v16bf v; ushort8 h[2]; };

__device__ inline unsigned short f2bf(float f) {
  unsigned int u = __float_as_uint(f);
  u += 0x7FFFu + ((u >> 16) & 1u);          // round-to-nearest-even
  return (unsigned short)(u >> 16);
}

#define WMMA_BF16(Af, Bf, Cv) \
  __builtin_amdgcn_wmma_f32_16x16x32_bf16(false, (Af).v, false, (Bf).v, (short)0, (Cv), false, false)

// ---------------------------------------------------------------------------
// Pack feat_w (fp32, KF x K1 row-major) into per-tile WMMA B layout, bf16,
// K zero-padded to KP. Layout: [kt][nt][lane][16 bf16] where lane&15 = n,
// (lane>>4)*8 = k-subgroup; elems 0..7 -> K=k0+g+0..7, elems 8..15 -> K=k0+16+g+0..7.
__global__ void pack_b_feat_kernel(const float* __restrict__ W,
                                   unsigned short* __restrict__ Bw) {
  int gid  = blockIdx.x * blockDim.x + threadIdx.x;
  int e    = gid & 15;
  int lane = (gid >> 4) & 31;
  int rest = gid >> 9;
  int nt   = rest % NTILES;
  int kt   = rest / NTILES;
  if (kt >= KTILES) return;
  int n = nt * 16 + (lane & 15);
  int g = (lane >> 4) * 8;
  int k = kt * 32 + ((e < 8) ? (g + e) : (16 + g + (e - 8)));
  float v = (k < KF) ? W[(size_t)k * K1 + n] : 0.f;
  Bw[gid] = f2bf(v);
}

// Pack cls_w (fp32, K1 x CC) into WMMA B layout with N padded 6 -> 16.
__global__ void pack_b_cls_kernel(const float* __restrict__ W,
                                  unsigned short* __restrict__ Bc) {
  int gid  = blockIdx.x * blockDim.x + threadIdx.x;
  int e    = gid & 15;
  int lane = (gid >> 4) & 31;
  int kt   = gid >> 9;
  if (kt >= KTILES_C) return;
  int n = lane & 15;
  int g = (lane >> 4) * 8;
  int k = kt * 32 + ((e < 8) ? (g + e) : (16 + g + (e - 8)));
  float v = (n < CC) ? W[(size_t)k * CC + n] : 0.f;
  Bc[gid] = f2bf(v);
}

// ---------------------------------------------------------------------------
// Hop-0 features: A0[row=(b,i,j)][0:768]=hidden[b,i], [768:1536]=hidden[b,j], rest 0.
__global__ void init_features_kernel(const float* __restrict__ hidden,
                                     unsigned short* __restrict__ A0) {
  int row = blockIdx.x;
  int b = row / (LL * LL), rem = row % (LL * LL);
  int i = rem / LL, j = rem % LL;
  const float* hi = hidden + ((size_t)b * LL + i) * HH;
  const float* hj = hidden + ((size_t)b * LL + j) * HH;
  unsigned short* dst = A0 + (size_t)row * KP;
  for (int col = threadIdx.x; col < KP; col += blockDim.x) {
    float v = 0.f;
    if (col < HH)            v = hi[col];
    else if (col < 2 * HH)   v = hj[col - HH];
    dst[col] = f2bf(v);
  }
}

// Hop-0 logits factorization: P[r,c] = hidden[r]·cls_w[0:H,c], Q = ·cls_w[H:2H,c]
__global__ void pq_kernel(const float* __restrict__ hidden,
                          const float* __restrict__ cls_w,
                          float* __restrict__ P, float* __restrict__ Q) {
  int gid = blockIdx.x * blockDim.x + threadIdx.x;
  if (gid >= BB * LL * CC) return;
  int c = gid % CC;
  int r = gid / CC;                     // [0, BB*LL)
  const float* h = hidden + (size_t)r * HH;
  float p = 0.f, q = 0.f;
  for (int k = 0; k < HH; ++k) {
    float hv = h[k];
    p += hv * cls_w[(size_t)k * CC + c];
    q += hv * cls_w[(size_t)(k + HH) * CC + c];
  }
  P[gid] = p; Q[gid] = q;
}

__global__ void logits0_kernel(const float* __restrict__ P,
                               const float* __restrict__ Q,
                               const float* __restrict__ cls_b,
                               float* __restrict__ logits) {
  int gid = blockIdx.x * blockDim.x + threadIdx.x;
  if (gid >= MROWS * CC) return;
  int c = gid % CC, row = gid / CC;
  int b = row / (LL * LL), rem = row % (LL * LL);
  int i = rem / LL, j = rem % LL;
  logits[gid] = P[(b * LL + i) * CC + c] + Q[(b * LL + j) * CC + c] + cls_b[c];
}

// ---------------------------------------------------------------------------
// pooled[b,r,c] = max(max_i lm[b,i,r,c], max_j lm[b,r,j,c]) with lm = probs*mask,
// masked entries contribute exactly 0 (matching jnp semantics).
__global__ void pool_kernel(const float* __restrict__ logits,
                            const float* __restrict__ amask,
                            float* __restrict__ pooled) {
  int b = blockIdx.x / LL, r = blockIdx.x % LL;
  int i = threadIdx.x;                  // 0..127
  float mm = amask[b * LL + r] * amask[b * LL + i];
  __shared__ float red[CC][LL];
  for (int c = 0; c < CC; ++c) {
    float v1 = (i <= r) ? logits[(((size_t)(b * LL + i)) * LL + r) * CC + c] * mm : 0.f;
    float v2 = (i >= r) ? logits[(((size_t)(b * LL + r)) * LL + i) * CC + c] * mm : 0.f;
    red[c][i] = fmaxf(v1, v2);
  }
  __syncthreads();
  for (int s = LL / 2; s > 0; s >>= 1) {
    if (i < s)
      for (int c = 0; c < CC; ++c)
        red[c][i] = fmaxf(red[c][i], red[c][i + s]);
    __syncthreads();
  }
  if (i == 0)
    for (int c = 0; c < CC; ++c)
      pooled[(b * LL + r) * CC + c] = red[c][0];
}

// Write the 18 concat columns (+ zero pad) of each row in place: cols
// [1536:1542)=pooled_i, [1542:1548)=pooled_j, [1548:1554)=probs, [1554:1568)=0.
__global__ void fill_extra_kernel(unsigned short* __restrict__ A,
                                  const float* __restrict__ pooled,
                                  const float* __restrict__ logits) {
  int row = blockIdx.x, t = threadIdx.x;   // t in [0,32)
  int b = row / (LL * LL), rem = row % (LL * LL);
  int i = rem / LL, j = rem % LL;
  float v = 0.f;
  if (t < CC)              v = pooled[(b * LL + i) * CC + t];
  else if (t < 2 * CC)     v = pooled[(b * LL + j) * CC + (t - CC)];
  else if (t < 3 * CC)     v = logits[(size_t)row * CC + (t - 2 * CC)];
  A[(size_t)row * KP + K1 + t] = f2bf(v);
}

// ---------------------------------------------------------------------------
// Main GEMM. Block = 4 waves; each wave owns a 32x64 output tile (two M-tiles
// x four N-tiles = 8 accumulators), so each B fragment feeds 2 WMMAs and each
// k-step runs 8 WMMAs on 6 fragment loads (0.75 loads/WMMA). Software-pipelined
// with even/odd fragment sets; launch_bounds(...,1) keeps both sets in VGPRs.
__global__ void __launch_bounds__(128, 1)
gemm_feat_kernel(const unsigned short* __restrict__ A,
                 const unsigned short* __restrict__ Bw,
                 const float* __restrict__ feat_b,
                 unsigned short* __restrict__ Aout) {
  const int lane = threadIdx.x & 31;
  const int wave = threadIdx.x >> 5;
  const int row0 = blockIdx.x * 32;
  const int n0   = blockIdx.y * 256 + wave * 64;
  const int m = lane & 15;
  const int g = (lane >> 4) * 8;
  v8f accA[4] = {};                       // rows row0..row0+15
  v8f accB[4] = {};                       // rows row0+16..row0+31
  const unsigned short* arow0 = A + (size_t)(row0 + m) * KP + g;
  const unsigned short* arow1 = arow0 + (size_t)16 * KP;
  const unsigned short* bbase = Bw + ((size_t)(n0 >> 4) * 32 + lane) * 16;
  const size_t bstride = (size_t)NTILES * 512;   // ushorts per k-step in Bw

  Frag a0A, a0B, a1A, a1B, b0[4], b1[4];
  a0A.h[0] = *(const ushort8*)(arow0);
  a0A.h[1] = *(const ushort8*)(arow0 + 16);
  a0B.h[0] = *(const ushort8*)(arow1);
  a0B.h[1] = *(const ushort8*)(arow1 + 16);
#pragma unroll
  for (int t = 0; t < 4; ++t) {
    b0[t].h[0] = *(const ushort8*)(bbase + t * 512);
    b0[t].h[1] = *(const ushort8*)(bbase + t * 512 + 8);
  }

  for (int kt = 0; kt + 2 <= KTILES - 1; kt += 2) {
    // stage next (kt+1) while computing kt
    const unsigned short* an0 = arow0 + (size_t)(kt + 1) * 32;
    const unsigned short* an1 = arow1 + (size_t)(kt + 1) * 32;
    const unsigned short* bn  = bbase + (size_t)(kt + 1) * bstride;
    a1A.h[0] = *(const ushort8*)(an0);
    a1A.h[1] = *(const ushort8*)(an0 + 16);
    a1B.h[0] = *(const ushort8*)(an1);
    a1B.h[1] = *(const ushort8*)(an1 + 16);
#pragma unroll
    for (int t = 0; t < 4; ++t) {
      b1[t].h[0] = *(const ushort8*)(bn + t * 512);
      b1[t].h[1] = *(const ushort8*)(bn + t * 512 + 8);
    }
    __builtin_prefetch(arow0 + (size_t)(kt + 4) * 32, 0, 3);
    __builtin_prefetch(arow1 + (size_t)(kt + 4) * 32, 0, 3);
#pragma unroll
    for (int t = 0; t < 4; ++t) {
      accA[t] = WMMA_BF16(a0A, b0[t], accA[t]);
      accB[t] = WMMA_BF16(a0B, b0[t], accB[t]);
    }

    // stage (kt+2) while computing kt+1
    const unsigned short* am0 = arow0 + (size_t)(kt + 2) * 32;
    const unsigned short* am1 = arow1 + (size_t)(kt + 2) * 32;
    const unsigned short* bm  = bbase + (size_t)(kt + 2) * bstride;
    a0A.h[0] = *(const ushort8*)(am0);
    a0A.h[1] = *(const ushort8*)(am0 + 16);
    a0B.h[0] = *(const ushort8*)(am1);
    a0B.h[1] = *(const ushort8*)(am1 + 16);
#pragma unroll
    for (int t = 0; t < 4; ++t) {
      b0[t].h[0] = *(const ushort8*)(bm + t * 512);
      b0[t].h[1] = *(const ushort8*)(bm + t * 512 + 8);
    }
#pragma unroll
    for (int t = 0; t < 4; ++t) {
      accA[t] = WMMA_BF16(a1A, b1[t], accA[t]);
      accB[t] = WMMA_BF16(a1B, b1[t], accB[t]);
    }
  }
  // final k-step (kt = KTILES-1 = 48, held in set 0)
#pragma unroll
  for (int t = 0; t < 4; ++t) {
    accA[t] = WMMA_BF16(a0A, b0[t], accA[t]);
    accB[t] = WMMA_BF16(a0B, b0[t], accB[t]);
  }

  const int nlo = lane & 15;
  const int mhi = (lane >> 4) * 8;
#pragma unroll
  for (int t = 0; t < 4; ++t) {
    int col = n0 + t * 16 + nlo;
    float bias = feat_b[col];
#pragma unroll
    for (int r = 0; r < 8; ++r) {
      int row = row0 + mhi + r;
      Aout[(size_t)row * KP + col]        = f2bf(accA[t][r] + bias);
      Aout[(size_t)(row + 16) * KP + col] = f2bf(accB[t][r] + bias);
    }
  }
}

// logits = features @ cls_w + cls_b  (N padded to one 16-wide tile, 6 valid)
__global__ void __launch_bounds__(32)
gemm_logits_kernel(const unsigned short* __restrict__ A,
                   const unsigned short* __restrict__ Bc,
                   const float* __restrict__ cls_b,
                   float* __restrict__ out) {
  const int lane = threadIdx.x;
  const int row0 = blockIdx.x * 16;
  const int m = lane & 15;
  const int g = (lane >> 4) * 8;
  v8f acc = {};
  const unsigned short* arow = A + (size_t)(row0 + m) * KP + g;

  Frag a0, a1, b0, b1;
  a0.h[0] = *(const ushort8*)(arow);
  a0.h[1] = *(const ushort8*)(arow + 16);
  b0.h[0] = *(const ushort8*)(Bc + (size_t)lane * 16);
  b0.h[1] = *(const ushort8*)(Bc + (size_t)lane * 16 + 8);

  // KTILES_C = 48 (even): pipeline pairs, last pair handled after the loop.
  for (int kt = 0; kt < KTILES_C - 2; kt += 2) {
    const unsigned short* an = arow + (size_t)(kt + 1) * 32;
    const unsigned short* bn = Bc + ((size_t)(kt + 1) * 32 + lane) * 16;
    a1.h[0] = *(const ushort8*)(an);
    a1.h[1] = *(const ushort8*)(an + 16);
    b1.h[0] = *(const ushort8*)(bn);
    b1.h[1] = *(const ushort8*)(bn + 8);
    acc = WMMA_BF16(a0, b0, acc);
    const unsigned short* an2 = arow + (size_t)(kt + 2) * 32;
    const unsigned short* bn2 = Bc + ((size_t)(kt + 2) * 32 + lane) * 16;
    a0.h[0] = *(const ushort8*)(an2);
    a0.h[1] = *(const ushort8*)(an2 + 16);
    b0.h[0] = *(const ushort8*)(bn2);
    b0.h[1] = *(const ushort8*)(bn2 + 8);
    acc = WMMA_BF16(a1, b1, acc);
  }
  {
    const int kt = KTILES_C - 2;            // 46 in set0; load 47 then finish
    const unsigned short* an = arow + (size_t)(kt + 1) * 32;
    const unsigned short* bn = Bc + ((size_t)(kt + 1) * 32 + lane) * 16;
    a1.h[0] = *(const ushort8*)(an);
    a1.h[1] = *(const ushort8*)(an + 16);
    b1.h[0] = *(const ushort8*)(bn);
    b1.h[1] = *(const ushort8*)(bn + 8);
    acc = WMMA_BF16(a0, b0, acc);
    acc = WMMA_BF16(a1, b1, acc);
  }

  const int n = lane & 15;
  if (n < CC) {
    float bias = cls_b[n];
    const int mhi = (lane >> 4) * 8;
#pragma unroll
    for (int r = 0; r < 8; ++r)
      out[(size_t)(row0 + mhi + r) * CC + n] = acc[r] + bias;
  }
}

// ---------------------------------------------------------------------------
extern "C" void kernel_launch(void* const* d_in, const int* in_sizes, int n_in,
                              void* d_out, int out_size, void* d_ws, size_t ws_size,
                              hipStream_t stream) {
  const float* hidden = (const float*)d_in[0];
  const float* amask  = (const float*)d_in[1];
  const float* cls_w  = (const float*)d_in[2];
  const float* cls_b  = (const float*)d_in[3];
  const float* feat_w = (const float*)d_in[4];
  const float* feat_b = (const float*)d_in[5];
  float* out = (float*)d_out;
  (void)in_sizes; (void)n_in; (void)out_size; (void)ws_size;

  char* ws = (char*)d_ws;
  size_t off = 0;
  auto carve = [&](size_t bytes) -> char* {
    char* p = ws + off;
    off = (off + bytes + 255) & ~(size_t)255;
    return p;
  };
  unsigned short* A0 = (unsigned short*)carve((size_t)MROWS * KP * 2);
  unsigned short* A1 = (unsigned short*)carve((size_t)MROWS * KP * 2);
  unsigned short* Bw = (unsigned short*)carve((size_t)KTILES * NTILES * 32 * 16 * 2);
  unsigned short* Bc = (unsigned short*)carve((size_t)KTILES_C * 32 * 16 * 2);
  float* logits = (float*)carve((size_t)MROWS * CC * 4);
  float* pooled = (float*)carve((size_t)BB * LL * CC * 4);
  float* P      = (float*)carve((size_t)BB * LL * CC * 4);
  float* Q      = (float*)carve((size_t)BB * LL * CC * 4);

  pack_b_feat_kernel<<<(KTILES * NTILES * 32 * 16) / 256, 256, 0, stream>>>(feat_w, Bw);
  pack_b_cls_kernel<<<(KTILES_C * 32 * 16) / 256, 256, 0, stream>>>(cls_w, Bc);
  init_features_kernel<<<MROWS, 256, 0, stream>>>(hidden, A0);
  pq_kernel<<<(BB * LL * CC + 255) / 256, 256, 0, stream>>>(hidden, cls_w, P, Q);
  logits0_kernel<<<(MROWS * CC) / 256, 256, 0, stream>>>(P, Q, cls_b, logits);

  unsigned short* Acur = A0;
  unsigned short* Anext = A1;
  for (int hop = 0; hop < 2; ++hop) {
    pool_kernel<<<BB * LL, LL, 0, stream>>>(logits, amask, pooled);
    fill_extra_kernel<<<MROWS, 32, 0, stream>>>(Acur, pooled, logits);
    dim3 gg(MROWS / 32, K1 / 256);
    gemm_feat_kernel<<<gg, 128, 0, stream>>>(Acur, Bw, feat_b, Anext);
    float* lo = (hop == 1) ? out : logits;
    gemm_logits_kernel<<<MROWS / 16, 32, 0, stream>>>(Anext, Bc, cls_b, lo);
    unsigned short* tmp = Acur; Acur = Anext; Anext = tmp;
  }
}
